// MultiheadAttentionRelPE_81990925680866
// MI455X (gfx1250) — matrix operand order repaired and checked by reading, not compiled, not run on hardware
//
#include <hip/hip_runtime.h>

// ---------------------------------------------------------------------------
// MultiheadAttention with relative PE bias — MI455X (gfx1250, wave32, WMMA)
//   EMBED=512, H=8, HD=64, N=L=2048, B=4
// Pipeline:
//   0) cvt_f32_f16_kernel x7 : one-shot f32->f16 of activations + weights
//   1) gemm_f16_kernel    x3 : pure-f16 WMMA GEMM (Qp, Kp natural; VpT [b][e][l])
//   2) attn_kernel            : flash attention, PE folded into QK^T chunk
//   3) gemm_f16_kernel    x1 : final projection, f32 output
// ---------------------------------------------------------------------------

#define EMBED 512
#define HDIM  64
#define NHEAD 8
#define NQ    2048
#define LK    2048
#define BB    4
#define TOK   (NQ * BB)   // 8192 token rows

typedef __attribute__((ext_vector_type(16))) _Float16 v16h;
typedef __attribute__((ext_vector_type(8)))  _Float16 v8h;
typedef __attribute__((ext_vector_type(2)))  _Float16 v2h;
typedef __attribute__((ext_vector_type(8)))  float    v8f;

__device__ inline v8f wmma_f16(v16h a, v16h b, v8f c) {
  // D = A(16x32 f16) * B(32x16 f16) + C(16x16 f32)
  return __builtin_amdgcn_wmma_f32_16x16x32_f16(
      /*neg_a=*/false, a, /*neg_b=*/false, b,
      /*c_mod=*/(short)0, c, /*reuse_a=*/false, /*reuse_b=*/false);
}

// A-matrix 16x32 f16 layout: lane l holds row M=l&15; element pair (2j,2j+1)
// holds K = ((j>>2)*16) + (l>>4)*8 + (j&3)*2  (and K+1).
__device__ inline int a_choff(int j, int kh) {
  return ((j >> 2) << 4) + (kh << 3) + ((j & 3) << 1);
}
// B operand as loaded in this file (A matches): lane l holds column N=l&15; element pair (2j,2j+1)
// holds K = (l>>4)*16 + 2j (and K+1): contiguous 16 K values per lane-half.

// Wave-local LDS producer->consumer ordering (cross-lane data through LDS).
__device__ inline void wave_lds_fence() {
  __builtin_amdgcn_wave_barrier();
  asm volatile("s_wait_dscnt 0" ::: "memory");
  __builtin_amdgcn_wave_barrier();
}

// ---------------------------------------------------------------------------
// Kernel 0: bulk f32 -> f16 conversion (bandwidth-trivial at 23.3 TB/s).
// n must be a multiple of 8 (all tensors here are multiples of 512).
// ---------------------------------------------------------------------------
__global__ __launch_bounds__(256)
void cvt_f32_f16_kernel(const float* __restrict__ src, _Float16* __restrict__ dst,
                        int n) {
  const int stride = gridDim.x * 256 * 8;
  for (int i = (blockIdx.x * 256 + threadIdx.x) * 8; i < n; i += stride) {
    v8f x = *(const v8f*)(src + i);
    v8h y;
#pragma unroll
    for (int j = 0; j < 8; ++j) y[j] = (_Float16)x[j];
    *(v8h*)(dst + i) = y;
  }
}

// ---------------------------------------------------------------------------
// Kernel 1: Y = A @ W^T + bias.  A:(T,512) f16, W:(512,512) f16 (row = out ch).
//   mode 0 : Y f16 natural rows Y[t][e]
//   mode 1 : Y f16 V-transposed Y[(b*EMBED+e)*LK + l],  t = l*BB + b
//   mode 2 : Y f32 natural rows (final output)
// Block: 256 thr (8 waves). Wave tile: 16 rows x 64 cols. WG tile: 128x64.
// Hot loop: f16 b128 loads + 4 WMMA, no conversions.
// ---------------------------------------------------------------------------
__global__ __launch_bounds__(256)
void gemm_f16_kernel(const _Float16* __restrict__ A, const _Float16* __restrict__ W,
                     const float* __restrict__ bias, void* __restrict__ Y,
                     int mode) {
  const int lane = threadIdx.x & 31;
  const int wave = threadIdx.x >> 5;
  const int m    = lane & 15;
  const int kh   = lane >> 4;
  const int rowBase = blockIdx.x * 128 + wave * 16;
  const int colBase = blockIdx.y * 64;

  v8f acc[4] = {};

  for (int k0 = 0; k0 < EMBED; k0 += 32) {
    // ---- A fragment (two 16B f16 loads) ----
    const _Float16* ar = A + (size_t)(rowBase + m) * EMBED + k0;
    v8h lo = *(const v8h*)(ar + kh * 16);
    v8h hi = *(const v8h*)(ar + kh * 16 + 8);
    v16h a;
#pragma unroll
    for (int i = 0; i < 8; ++i) {
      a[i]     = lo[i];
      a[8 + i] = hi[i];
    }
#pragma unroll
    for (int t = 0; t < 4; ++t) {
      // ---- B fragment: W row = output channel; contiguous 16 K per lane ----
      const _Float16* wr = W + (size_t)(colBase + t * 16 + m) * EMBED + k0;
      v16h b = *(const v16h*)(wr + kh * 16);
      acc[t] = wmma_f16(a, b, acc[t]);
    }
  }

  // ---- store (C layout: lane col = m, rows r + 8*kh) ----
#pragma unroll
  for (int t = 0; t < 4; ++t) {
    const int col = colBase + t * 16 + m;
    const float bv = bias[col];
#pragma unroll
    for (int r = 0; r < 8; ++r) {
      const int row = rowBase + r + 8 * kh;
      const float val = acc[t][r] + bv;
      if (mode == 0) {
        ((_Float16*)Y)[(size_t)row * EMBED + col] = (_Float16)val;
      } else if (mode == 1) {
        const int l = row >> 2;   // t = l*BB + b, BB==4
        const int b = row & 3;
        ((_Float16*)Y)[((size_t)b * EMBED + col) * LK + l] = (_Float16)val;
      } else {
        ((float*)Y)[(size_t)row * EMBED + col] = val;
      }
    }
  }
}

// ---------------------------------------------------------------------------
// Kernel 2: flash attention.
//   S = (Q'·K'^T) / sqrt(HD), with Q'=[Qh | pe1,0...], K'=[Kh | pe2,0...]
//   (head dim padded 64 -> 96; pe bias becomes one extra WMMA chunk).
// Block: 256 thr = 8 waves; each wave owns 16 query rows of one (b,h).
// Grid: (B*H, NQ/128).
// ---------------------------------------------------------------------------
__global__ __launch_bounds__(256)
void attn_kernel(const _Float16* __restrict__ Qp, const _Float16* __restrict__ Kp,
                 const _Float16* __restrict__ VpT,
                 const float* __restrict__ pe1, const float* __restrict__ pe2,
                 _Float16* __restrict__ Oh) {
  __shared__ _Float16 Plds[8][16 * 32];   // per-wave 16x32 P tile (8 KB)

  const int lane = threadIdx.x & 31;
  const int wave = threadIdx.x >> 5;
  const int n16  = lane & 15;
  const int kh   = lane >> 4;

  const int bh = blockIdx.x;          // 0..31
  const int b  = bh >> 3;
  const int h  = bh & (NHEAD - 1);
  const int qBase = (blockIdx.y * 8 + wave) * 16;

  // ---- Q' A-fragments: chunks 0,1 = head channels; chunk 2 = pe1 (padded) ----
  v16h qa[3];
  {
    const _Float16* qr = Qp + ((size_t)(qBase + n16) * BB + b) * EMBED + h * HDIM;
#pragma unroll
    for (int kc = 0; kc < 2; ++kc) {
      v8h lo = *(const v8h*)(qr + kc * 32 + kh * 16);
      v8h hi = *(const v8h*)(qr + kc * 32 + kh * 16 + 8);
#pragma unroll
      for (int i = 0; i < 8; ++i) {
        qa[kc][i]     = lo[i];
        qa[kc][8 + i] = hi[i];
      }
    }
    v16h pa;
#pragma unroll
    for (int i = 0; i < 16; ++i) pa[i] = (_Float16)0.f;
    if (kh == 0) {  // channels 64..66 live in lane-half 0, j=0..1
      const float* p1 = pe1 + ((size_t)b * NQ + (qBase + n16)) * 3;
      pa[0] = (_Float16)p1[0];
      pa[1] = (_Float16)p1[1];
      pa[2] = (_Float16)p1[2];
    }
    qa[2] = pa;
  }

  v8f o[4] = {};
  float mrun[8], lrun[8];
#pragma unroll
  for (int r = 0; r < 8; ++r) { mrun[r] = -1e30f; lrun[r] = 0.f; }

  const float scale = 0.125f;   // 1/sqrt(64)

  for (int kb = 0; kb < LK; kb += 32) {
    v8f s[2] = {};
    // ---- two 16-key score tiles ----
#pragma unroll
    for (int st = 0; st < 2; ++st) {
      const int key = kb + st * 16 + n16;   // this lane's key column
      const _Float16* kr = Kp + ((size_t)key * BB + b) * EMBED + h * HDIM;
#pragma unroll
      for (int kc = 0; kc < 2; ++kc) {
        v16h bf = *(const v16h*)(kr + kc * 32 + kh * 16);   // contiguous 16 ch
        s[st] = wmma_f16(qa[kc], bf, s[st]);
      }
      // pe chunk (rows 64..66 of K' -> lane-half 0, j=0..1)
      v16h pb;
#pragma unroll
      for (int i = 0; i < 16; ++i) pb[i] = (_Float16)0.f;
      if (kh == 0) {
        const float* p2 = pe2 + (size_t)key * 3;
        pb[0] = (_Float16)p2[0];
        pb[1] = (_Float16)p2[1];
        pb[2] = (_Float16)p2[2];
      }
      s[st] = wmma_f16(qa[2], pb, s[st]);
    }

    // ---- online softmax (row stats across 16-lane half-groups) ----
#pragma unroll
    for (int r = 0; r < 8; ++r) {
      float v0 = s[0][r] * scale;
      float v1 = s[1][r] * scale;
      float mx = fmaxf(v0, v1);
#pragma unroll
      for (int off = 8; off >= 1; off >>= 1)
        mx = fmaxf(mx, __shfl_xor(mx, off, 32));
      const float mnew  = fmaxf(mrun[r], mx);
      const float alpha = __expf(mrun[r] - mnew);
      const float p0 = __expf(v0 - mnew);
      const float p1 = __expf(v1 - mnew);
      float rsum = p0 + p1;
#pragma unroll
      for (int off = 8; off >= 1; off >>= 1)
        rsum += __shfl_xor(rsum, off, 32);
      lrun[r] = lrun[r] * alpha + rsum;
      mrun[r] = mnew;
#pragma unroll
      for (int t = 0; t < 4; ++t) o[t][r] *= alpha;
      // stash exp'd probs: P[row][key], row = r + 8*kh, cols n16 / 16+n16
      const int row = r + 8 * kh;
      Plds[wave][row * 32 + n16]      = (_Float16)p0;
      Plds[wave][row * 32 + 16 + n16] = (_Float16)p1;
    }

    wave_lds_fence();

    // ---- reload P as A-fragment (C layout -> A layout transpose via LDS) ----
    v16h pa;
#pragma unroll
    for (int j = 0; j < 8; ++j) {
      const int key = kh * 16 + 2 * j;
      const v2h pr = *(const v2h*)&Plds[wave][n16 * 32 + key];
      pa[2 * j]     = pr.x;
      pa[2 * j + 1] = pr.y;
    }

    // ---- O += P(16x32) x V(32x64), V transposed so K is contiguous ----
#pragma unroll
    for (int t = 0; t < 4; ++t) {
      const int ch = h * HDIM + t * 16 + n16;
      const _Float16* vr = VpT + ((size_t)b * EMBED + ch) * LK + kb + kh * 16;
      v16h vb = *(const v16h*)vr;   // rows kh*16 .. kh*16+15 of this K-chunk
      o[t] = wmma_f16(pa, vb, o[t]);
    }
  }

  // ---- normalize and store head output (natural f16 rows) ----
#pragma unroll
  for (int t = 0; t < 4; ++t) {
#pragma unroll
    for (int r = 0; r < 8; ++r) {
      const int qrow = qBase + r + 8 * kh;
      const float val = o[t][r] / lrun[r];
      Oh[((size_t)qrow * BB + b) * EMBED + h * HDIM + t * 16 + n16] = (_Float16)val;
    }
  }
}

// ---------------------------------------------------------------------------
extern "C" void kernel_launch(void* const* d_in, const int* in_sizes, int n_in,
                              void* d_out, int out_size, void* d_ws, size_t ws_size,
                              hipStream_t stream) {
  (void)in_sizes; (void)n_in; (void)out_size; (void)ws_size;

  const float* q   = (const float*)d_in[0];
  const float* k   = (const float*)d_in[1];
  const float* v   = (const float*)d_in[2];
  const float* pe1 = (const float*)d_in[3];
  const float* pe2 = (const float*)d_in[4];
  const float* Wq  = (const float*)d_in[5];
  const float* bq  = (const float*)d_in[6];
  const float* Wk  = (const float*)d_in[7];
  const float* bk  = (const float*)d_in[8];
  const float* Wv  = (const float*)d_in[9];
  const float* bv  = (const float*)d_in[10];
  const float* Wo  = (const float*)d_in[11];
  const float* bo  = (const float*)d_in[12];
  float* out = (float*)d_out;

  _Float16* ws = (_Float16*)d_ws;
  const size_t SZ  = (size_t)TOK * EMBED;       // 4M f16 elems (8 MB)
  const size_t WSZ = (size_t)EMBED * EMBED;     // 256K f16 elems (0.5 MB)
  _Float16* Xq  = ws;                           // f16 activations
  _Float16* Xk  = ws + SZ;
  _Float16* Xv  = ws + 2 * SZ;
  _Float16* Qp  = ws + 3 * SZ;                  // projected f16
  _Float16* Kp  = ws + 4 * SZ;
  _Float16* VpT = ws + 5 * SZ;                  // [b][e][l]
  _Float16* Oh  = ws + 6 * SZ;
  _Float16* Whq = ws + 7 * SZ;                  // f16 weights
  _Float16* Whk = Whq + WSZ;
  _Float16* Whv = Whk + WSZ;
  _Float16* Who = Whv + WSZ;

  dim3 blk(256);

  // ---- 0) one-shot f32 -> f16 conversions ----
  const int nAct = (int)SZ;     // 4,194,304
  const int nW   = (int)WSZ;    // 262,144
  dim3 gAct(nAct / (256 * 8));  // 2048 blocks
  dim3 gW(nW / (256 * 8));      // 128 blocks
  cvt_f32_f16_kernel<<<gAct, blk, 0, stream>>>(q, Xq, nAct);
  cvt_f32_f16_kernel<<<gAct, blk, 0, stream>>>(k, Xk, nAct);
  cvt_f32_f16_kernel<<<gAct, blk, 0, stream>>>(v, Xv, nAct);
  cvt_f32_f16_kernel<<<gW, blk, 0, stream>>>(Wq, Whq, nW);
  cvt_f32_f16_kernel<<<gW, blk, 0, stream>>>(Wk, Whk, nW);
  cvt_f32_f16_kernel<<<gW, blk, 0, stream>>>(Wv, Whv, nW);
  cvt_f32_f16_kernel<<<gW, blk, 0, stream>>>(Wo, Who, nW);

  // ---- 1) input projections (pure f16 WMMA GEMM) ----
  dim3 gProj(TOK / 128, EMBED / 64);            // 64 x 8
  gemm_f16_kernel<<<gProj, blk, 0, stream>>>(Xq, Whq, bq, Qp, 0);
  gemm_f16_kernel<<<gProj, blk, 0, stream>>>(Xk, Whk, bk, Kp, 0);
  gemm_f16_kernel<<<gProj, blk, 0, stream>>>(Xv, Whv, bv, VpT, 1);

  // ---- 2) flash attention ----
  dim3 gAttn(BB * NHEAD, NQ / 128);             // 32 x 16
  attn_kernel<<<gAttn, blk, 0, stream>>>(Qp, Kp, VpT, pe1, pe2, Oh);

  // ---- 3) output projection (f32 result) ----
  gemm_f16_kernel<<<gProj, blk, 0, stream>>>(Oh, Who, bo, out, 2);
}
